// VQQuantizer_46488726012198
// MI455X (gfx1250) — matrix operand order repaired and checked by reading, not compiled, not run on hardware
//
#include <hip/hip_runtime.h>

typedef __attribute__((ext_vector_type(16))) _Float16 v16h;
typedef __attribute__((ext_vector_type(8)))  float    v8f;

#define N_CODES 16384
#define DIM     64
#define NQ      32768                 // 8*32*32 spatial * K=4 groups
#define SCALE   16384.0f              // lift emb into fp16 normal range
#define SLICES  4
#define CODES_PER_SLICE (N_CODES / SLICES)
#define CHUNK   256                   // codes staged in LDS per step
#define NCHUNK  (CODES_PER_SLICE / CHUNK)
#define WAVES_PER_BLOCK 8
#define ROWS_PER_WAVE 32              // 2 M-tiles per wave

// dynamic LDS layout (bytes): 2 x (256 codes x 64 halves) + 2 x 256 floats
#define LDS_B_BYTES   (CHUNK * DIM * 2)          // 32768 per buffer
#define LDS_BIAS_OFF  (2 * LDS_B_BYTES)          // 65536
#define LDS_TOTAL     (LDS_BIAS_OFF + 2 * CHUNK * 4)

// workspace layout (bytes)
#define WS_BIAS_OFF  (N_CODES * DIM * 2)         // after f16 emb (2 MB)
#define WS_PVAL_OFF  (WS_BIAS_OFF + N_CODES * 4)
#define WS_PIDX_OFF  (WS_PVAL_OFF + NQ * SLICES * 4)

// output layout (floats)
#define OUT_ZQ_N     2097152                     // 8*256*32*32
#define OUT_IDX_OFF  OUT_ZQ_N
#define OUT_LOSS_OFF (OUT_ZQ_N + NQ)

// ---- CDNA5 async-to-LDS helpers (ASYNCcnt path, CDNA5 ISA) ----
__device__ __forceinline__ void async_b128(unsigned lds, const void* g) {
  asm volatile("global_load_async_to_lds_b128 %0, %1, off" :: "v"(lds), "v"(g) : "memory");
}
__device__ __forceinline__ void async_b32(unsigned lds, const void* g) {
  asm volatile("global_load_async_to_lds_b32 %0, %1, off" :: "v"(lds), "v"(g) : "memory");
}
__device__ __forceinline__ void wait_async0() {
  asm volatile("s_wait_asynccnt 0x0" ::: "memory");
}

// ---- prologue: emb fp32 -> scaled fp16 + fp32 negbias = -0.5*S*||e||^2 ----
__global__ void vq_prep(const float* __restrict__ emb,
                        _Float16* __restrict__ embh,
                        float* __restrict__ negb) {
  int m = blockIdx.x * blockDim.x + threadIdx.x;
  if (m >= N_CODES) return;
  const float* row = emb + m * DIM;
  _Float16* orow = embh + m * DIM;
  float s = 0.f;
  #pragma unroll
  for (int d = 0; d < DIM; ++d) {
    float e = row[d];
    s += e * e;
    orow[d] = (_Float16)(e * SCALE);
  }
  negb[m] = -0.5f * SCALE * s;
}

__global__ void vq_init(float* out) { out[OUT_LOSS_OFF] = 0.f; }

// ---- main: fused GEMM (WMMA f16->f32) + running argmax, async pipelined ----
__global__ __launch_bounds__(256) void vq_argmax(
    const float* __restrict__ z, const _Float16* __restrict__ embh,
    const float* __restrict__ negb,
    float* __restrict__ pval, int* __restrict__ pidx) {
  extern __shared__ __align__(32) char smem[];
  _Float16* tileB = (_Float16*)smem;                 // [2][CHUNK*DIM]
  float*    biasS = (float*)(smem + LDS_BIAS_OFF);   // [2][CHUNK]

  const int tid  = threadIdx.x;
  const int wave = tid >> 5;
  const int lane = tid & 31;
  const int row  = lane & 15;   // A row & B/C column
  const int hsel = lane >> 4;   // K half select (A/B), +8 rows (C/D)

  const int slice = blockIdx.y;
  const int qbase = (blockIdx.x * WAVES_PER_BLOCK + wave) * ROWS_PER_WAVE;
  const int kgrp  = qbase >> 13;       // which of 4 channel sub-groups
  const int n0    = qbase & 8191;
  const int b_i   = n0 >> 10;
  const int p0    = n0 & 1023;
  const int zbase = b_i * 262144 + p0 + row;  // z is (b, c, 32, 32) fp32

  // A fragments, 2 M-tiles (rows qbase..+15 and qbase+16..+31).
  // A layout (matches B): lane<16 -> K {0..15}; lane>=16 -> K {16..31}.
  v16h a0, a1, a2, a3;
  #pragma unroll
  for (int t = 0; t < 16; ++t) {
    int db = t + (hsel << 4);                             // 0..31
    a0[t] = (_Float16)z[zbase +      (((db      * 4) + kgrp) << 10)];
    a1[t] = (_Float16)z[zbase +      ((((db+32) * 4) + kgrp) << 10)];
    a2[t] = (_Float16)z[zbase + 16 + (((db      * 4) + kgrp) << 10)];
    a3[t] = (_Float16)z[zbase + 16 + ((((db+32) * 4) + kgrp) << 10)];
  }

  float bv0[8], bv1[8];
  int   bi0[8], bi1[8];
  #pragma unroll
  for (int r = 0; r < 8; ++r) {
    bv0[r] = bv1[r] = -3.0e38f;
    bi0[r] = bi1[r] = 0;
  }

  const int slice_base = slice * CODES_PER_SLICE;

  // stage chunk 0 into buffer 0 (async -> LDS)
  {
    const char* gsrc = (const char*)(embh + slice_base * DIM);
    unsigned ldsB = (unsigned)(uintptr_t)tileB;
    #pragma unroll
    for (int j = 0; j < LDS_B_BYTES / 16 / 256; ++j) {
      int off = (tid + 256 * j) * 16;
      async_b128(ldsB + off, gsrc + off);
    }
    async_b32((unsigned)(uintptr_t)(biasS + tid), negb + slice_base + tid);
  }
  wait_async0();
  __syncthreads();

  for (int chunk = 0; chunk < NCHUNK; ++chunk) {
    const int cb = chunk & 1;
    const int cbase = slice_base + chunk * CHUNK;

    // kick off async staging of the NEXT chunk into the other buffer
    if (chunk + 1 < NCHUNK) {
      const int nb2 = cb ^ 1;
      const char* gsrc = (const char*)(embh + (cbase + CHUNK) * DIM);
      unsigned ldsB = (unsigned)(uintptr_t)(tileB + nb2 * (CHUNK * DIM));
      #pragma unroll
      for (int j = 0; j < LDS_B_BYTES / 16 / 256; ++j) {
        int off = (tid + 256 * j) * 16;
        async_b128(ldsB + off, gsrc + off);
      }
      async_b32((unsigned)(uintptr_t)(biasS + nb2 * CHUNK + tid),
                negb + cbase + CHUNK + tid);
    }

    const _Float16* curB = tileB + cb * (CHUNK * DIM);
    const float*    curN = biasS + cb * CHUNK;

    for (int t16 = 0; t16 < CHUNK / 16; ++t16) {
      const int cl = (t16 << 4) + row;               // this lane's code column
      const _Float16* bp = curB + cl * DIM + (hsel << 4);
      v16h b0 = *(const v16h*)(bp);                  // K 0..31 span
      v16h b1 = *(const v16h*)(bp + 32);             // K 32..63 span
      float nb = curN[cl];                           // already negated bias
      v8f c0, c1;
      #pragma unroll
      for (int r = 0; r < 8; ++r) { c0[r] = nb; c1[r] = nb; }
      // interleave the two accumulators: no back-to-back RAW on C
      c0 = __builtin_amdgcn_wmma_f32_16x16x32_f16(false, a0, false, b0,
                                                  (short)0, c0, false, false);
      c1 = __builtin_amdgcn_wmma_f32_16x16x32_f16(false, a2, false, b0,
                                                  (short)0, c1, false, false);
      c0 = __builtin_amdgcn_wmma_f32_16x16x32_f16(false, a1, false, b1,
                                                  (short)0, c0, false, false);
      c1 = __builtin_amdgcn_wmma_f32_16x16x32_f16(false, a3, false, b1,
                                                  (short)0, c1, false, false);
      const int code = cbase + cl;
      #pragma unroll
      for (int r = 0; r < 8; ++r) {
        if (c0[r] > bv0[r]) { bv0[r] = c0[r]; bi0[r] = code; }
        if (c1[r] > bv1[r]) { bv1[r] = c1[r]; bi1[r] = code; }
      }
    }

    if (chunk + 1 < NCHUNK) wait_async0();   // our async copies done
    __syncthreads();                         // everyone done reading/writing
  }

  // argmax across the 16 columns held by each 16-lane half (prefer low idx)
  #pragma unroll
  for (int m = 8; m >= 1; m >>= 1) {
    #pragma unroll
    for (int r = 0; r < 8; ++r) {
      float ov = __shfl_xor(bv0[r], m, 16);
      int   oi = __shfl_xor(bi0[r], m, 16);
      if (ov > bv0[r] || (ov == bv0[r] && oi < bi0[r])) { bv0[r] = ov; bi0[r] = oi; }
      ov = __shfl_xor(bv1[r], m, 16);
      oi = __shfl_xor(bi1[r], m, 16);
      if (ov > bv1[r] || (ov == bv1[r] && oi < bi1[r])) { bv1[r] = ov; bi1[r] = oi; }
    }
  }
  if (row == 0) {
    int q0 = qbase + (hsel << 3);      // rows of C tile 0 held by this lane
    #pragma unroll
    for (int r = 0; r < 8; ++r) {
      pval[(q0 + r) * SLICES + slice] = bv0[r];
      pidx[(q0 + r) * SLICES + slice] = bi0[r];
      pval[(q0 + 16 + r) * SLICES + slice] = bv1[r];
      pidx[(q0 + 16 + r) * SLICES + slice] = bi1[r];
    }
  }
}

// ---- epilogue: merge slices, gather fp32 codes, write z_q/idx, loss ----
__global__ __launch_bounds__(256) void vq_combine(
    const float* __restrict__ z, const float* __restrict__ emb,
    const float* __restrict__ pval, const int* __restrict__ pidx,
    float* __restrict__ out) {
  __shared__ float red[256];
  int q = blockIdx.x * 256 + threadIdx.x;
  float bv = -3.0e38f; int bi = 0x7fffffff;
  #pragma unroll
  for (int s = 0; s < SLICES; ++s) {
    float v = pval[q * SLICES + s];
    int   i = pidx[q * SLICES + s];
    if (v > bv || (v == bv && i < bi)) { bv = v; bi = i; }
  }
  int k = q >> 13, n = q & 8191, b_i = n >> 10, p = n & 1023;
  out[OUT_IDX_OFF + b_i * 4096 + (k << 10) + p] = (float)bi;
  const float* er = emb + bi * DIM;
  int zo = b_i * 262144 + p;
  float sum = 0.f;
  #pragma unroll
  for (int d = 0; d < DIM; ++d) {
    float e = er[d];
    int o = zo + (((d << 2) + k) << 10);   // (b, c=d*4+k, h, w)
    float diff = e - z[o];
    sum += diff * diff;
    out[o] = e;                            // straight-through z_q == quantized
  }
  red[threadIdx.x] = sum;
  __syncthreads();
  for (int s = 128; s > 0; s >>= 1) {
    if (threadIdx.x < s) red[threadIdx.x] += red[threadIdx.x + s];
    __syncthreads();
  }
  if (threadIdx.x == 0)
    atomicAdd(out + OUT_LOSS_OFF, red[0] * (1.25f / 2097152.0f)); // (1+beta)*MSE
}

extern "C" void kernel_launch(void* const* d_in, const int* in_sizes, int n_in,
                              void* d_out, int out_size, void* d_ws, size_t ws_size,
                              hipStream_t stream) {
  (void)in_sizes; (void)n_in; (void)out_size; (void)ws_size;
  const float* z   = (const float*)d_in[0];
  const float* emb = (const float*)d_in[1];
  float* out = (float*)d_out;
  char* ws = (char*)d_ws;
  _Float16* embh = (_Float16*)ws;
  float* negb = (float*)(ws + WS_BIAS_OFF);
  float* pval = (float*)(ws + WS_PVAL_OFF);
  int*   pidx = (int*)(ws + WS_PIDX_OFF);

  vq_prep<<<N_CODES / 256, 256, 0, stream>>>(emb, embh, negb);
  vq_init<<<1, 1, 0, stream>>>(out);
  vq_argmax<<<dim3(NQ / ROWS_PER_WAVE / WAVES_PER_BLOCK, SLICES), 256,
              LDS_TOTAL, stream>>>(z, embh, negb, pval, pidx);
  vq_combine<<<NQ / 256, 256, 0, stream>>>(z, emb, pval, pidx, out);
}
